// triAMPh_89000312308145
// MI455X (gfx1250) — compile-verified, matrix operand order checked by reading, not run on hardware
//
#include <hip/hip_runtime.h>
#include <math.h>

typedef __attribute__((ext_vector_type(16))) _Float16 v16h;
typedef __attribute__((ext_vector_type(8)))  float    v8f;

#define NN   50000
#define EE   800000
#define EPN  200000
#define HH   4
#define DD   32
#define HIDN 128
#define HANIN 256

__device__ __forceinline__ unsigned pack2h(float a, float b) {
    union { _Float16 h[2]; unsigned u; } p;
    p.h[0] = (_Float16)a;
    p.h[1] = (_Float16)b;
    return p.u;
}

// ---------------------------------------------------------------------------
// generic fill
// ---------------------------------------------------------------------------
__global__ void han_fill_u32(unsigned* __restrict__ p, unsigned v, long long n) {
    long long i = (long long)blockIdx.x * blockDim.x + threadIdx.x;
    if (i < n) p[i] = v;
}

// ---------------------------------------------------------------------------
// WMMA GEMM: C[M,Nc] = A[M,K] @ B[K,Nc] + bias.
// Block: 256 threads (8 waves). Block tile 64x64; each wave -> 16x32 (2 frags).
// K multiple of 32, Nc multiple of 64. Branch-free f16 staging.
// ---------------------------------------------------------------------------
__global__ void han_gemm_wmma(const float* __restrict__ A, const float* __restrict__ B,
                              const float* __restrict__ bias, float* __restrict__ C,
                              int M, int K, int Nc) {
    __shared__ _Float16 As[64 * 34];
    __shared__ _Float16 Bs[64 * 34];
    const int tid  = threadIdx.x;
    const int lane = tid & 31;
    const int wave = tid >> 5;
    const int m0 = blockIdx.x * 64;
    const int n0 = blockIdx.y * 64;
    const int mt = (wave >> 1) * 16;
    const int nt = (wave & 1) * 32;
    const int lo = lane >> 4;
    const int ml = lane & 15;

    v8f c0 = {}; v8f c1 = {};

    const int ar  = tid >> 2;        // A tile row 0..63
    const int acg = (tid & 3) * 8;   // A tile col group (multiple of 8)
    const int bk  = tid >> 3;        // B tile k 0..31
    const int bng = (tid & 7) * 8;   // B tile n group

    const int   arow = m0 + ar;
    const float amsk = (arow < M) ? 1.f : 0.f;
    const int   arowc = (arow < M) ? arow : (M - 1);
    const float* aprow = A + (size_t)arowc * K + acg;
    const float* bprow = B + (size_t)bk * Nc + n0 + bng;

    for (int k0 = 0; k0 < K; k0 += 32) {
        {   // stage A (64 x 32) as f16: b128 loads, packed u32 LDS stores
            const float* ap = aprow + k0;
            float4 x0 = *(const float4*)(ap);
            float4 x1 = *(const float4*)(ap + 4);
            __builtin_prefetch(ap + 32, 0, 3);
            unsigned* q = (unsigned*)&As[ar * 34 + acg];
            q[0] = pack2h(x0.x * amsk, x0.y * amsk);
            q[1] = pack2h(x0.z * amsk, x0.w * amsk);
            q[2] = pack2h(x1.x * amsk, x1.y * amsk);
            q[3] = pack2h(x1.z * amsk, x1.w * amsk);
        }
        {   // stage B (32 x 64) as f16, stored N-major
            const float* bp = bprow + (size_t)k0 * Nc;
            float4 y0 = *(const float4*)(bp);
            float4 y1 = *(const float4*)(bp + 4);
            __builtin_prefetch(bp + (size_t)32 * Nc, 0, 3);
            Bs[(bng + 0) * 34 + bk] = (_Float16)y0.x;
            Bs[(bng + 1) * 34 + bk] = (_Float16)y0.y;
            Bs[(bng + 2) * 34 + bk] = (_Float16)y0.z;
            Bs[(bng + 3) * 34 + bk] = (_Float16)y0.w;
            Bs[(bng + 4) * 34 + bk] = (_Float16)y1.x;
            Bs[(bng + 5) * 34 + bk] = (_Float16)y1.y;
            Bs[(bng + 6) * 34 + bk] = (_Float16)y1.z;
            Bs[(bng + 7) * 34 + bk] = (_Float16)y1.w;
        }
        __syncthreads();

        union { v16h v; _Float16 h[16]; } af, bf0, bf1;
        #pragma unroll
        for (int e = 0; e < 16; ++e) {
            int kk = ((e >> 3) << 4) + (lo << 3) + (e & 7);
            af.h[e]  = As[(mt + ml) * 34 + kk];
            bf0.h[e] = Bs[(nt + ml) * 34 + kk];
            bf1.h[e] = Bs[(nt + 16 + ml) * 34 + kk];
        }
        c0 = __builtin_amdgcn_wmma_f32_16x16x32_f16(false, af.v, false, bf0.v, (short)0, c0, false, false);
        c1 = __builtin_amdgcn_wmma_f32_16x16x32_f16(false, af.v, false, bf1.v, (short)0, c1, false, false);
        __syncthreads();
    }

    // epilogue: bias hoisted out of the predicated row loop
    const int col = n0 + nt + ml;
    const float bv0 = bias[col];
    const float bv1 = bias[col + 16];
    float* crow = C + (size_t)(m0 + mt + 8 * lo) * Nc + col;
    #pragma unroll
    for (int r = 0; r < 8; ++r) {
        int row = m0 + mt + r + 8 * lo;
        if (row < M) {
            crow[0]  = c0[r] + bv0;
            crow[16] = c1[r] + bv1;
        }
        crow += Nc;
    }
}

// ---------------------------------------------------------------------------
// Edge logits + segment-max (one wave per edge; lanes = D=32; loop heads)
// ---------------------------------------------------------------------------
__global__ void han_edge_logits(const float* __restrict__ FS, const float* __restrict__ FD,
                                const float* __restrict__ attn,
                                const int* __restrict__ src, const int* __restrict__ dst,
                                float* __restrict__ logits, unsigned* __restrict__ maxb, int E) {
    int gw   = (int)(((long long)blockIdx.x * blockDim.x + threadIdx.x) >> 5);
    int lane = threadIdx.x & 31;
    if (gw >= E) return;
    int s = src[gw], d = dst[gw];
    const float* fsp = FS + (size_t)s * HIDN;
    const float* fdp = FD + (size_t)d * HIDN;
    #pragma unroll
    for (int h = 0; h < HH; ++h) {
        int j = h * DD + lane;
        float x  = fsp[j] + fdp[j];
        float lr = x > 0.f ? x : 0.2f * x;
        float p  = lr * attn[j];
        #pragma unroll
        for (int off = 16; off > 0; off >>= 1) p += __shfl_xor(p, off, 32);
        if (lane == 0) {
            logits[(size_t)gw * HH + h] = p;
            unsigned u = __float_as_uint(p);
            u = (u & 0x80000000u) ? ~u : (u | 0x80000000u);   // monotonic encode
            atomicMax(&maxb[(size_t)d * HH + h], u);
        }
    }
}

// ---------------------------------------------------------------------------
// exp(logit - max[dst]) in place + segment-sum
// ---------------------------------------------------------------------------
__global__ void han_edge_exp(float* __restrict__ logits, const int* __restrict__ dst,
                             const unsigned* __restrict__ maxb, float* __restrict__ sumb, int E) {
    int t = blockIdx.x * blockDim.x + threadIdx.x;
    if (t >= E * HH) return;
    int e = t >> 2, h = t & 3;
    int d = dst[e];
    unsigned u = maxb[(size_t)d * HH + h];
    float m = (u & 0x80000000u) ? __uint_as_float(u ^ 0x80000000u) : __uint_as_float(~u);
    float ex = __expf(logits[t] - m);
    logits[t] = ex;
    atomicAdd(&sumb[(size_t)d * HH + h], ex);
}

// ---------------------------------------------------------------------------
// alpha-weighted scatter: Z[dst] += (ex/sum[dst]) * FS[src]   (per head slice)
// ---------------------------------------------------------------------------
__global__ void han_edge_scatter(const float* __restrict__ FS, const float* __restrict__ ex,
                                 const float* __restrict__ sumb,
                                 const int* __restrict__ src, const int* __restrict__ dst,
                                 float* __restrict__ Z, int E) {
    long long t = (long long)blockIdx.x * blockDim.x + threadIdx.x;
    if (t >= (long long)E * HIDN) return;
    int e = (int)(t >> 7);
    int j = (int)(t & 127);
    int h = j >> 5;
    int s = src[e], d = dst[e];
    float alpha = ex[(size_t)e * HH + h] / sumb[(size_t)d * HH + h];
    atomicAdd(&Z[(size_t)d * HIDN + j], alpha * FS[(size_t)s * HIDN + j]);
}

__global__ void han_relu_inplace(float* __restrict__ p, long long n) {
    long long i = (long long)blockIdx.x * blockDim.x + threadIdx.x;
    if (i < n) p[i] = fmaxf(p[i], 0.f);
}

// ---------------------------------------------------------------------------
// Semantic attention score: wout += sum_n tanh(z[n]@Wa + ba) @ Wo
// one wave per node; Wa cached in LDS as f16
// ---------------------------------------------------------------------------
__global__ void han_sem_score(const float* __restrict__ Z, const float* __restrict__ Wa,
                              const float* __restrict__ ba, const float* __restrict__ Wo,
                              float* __restrict__ wout, int n_nodes) {
    __shared__ _Float16 Ws[128 * 128];
    __shared__ float bsh[128];
    __shared__ float wos[128];
    int tid = threadIdx.x;
    for (int i = tid * 4; i < 128 * 128; i += 256 * 4) {
        float4 v = *(const float4*)(Wa + i);
        unsigned* q = (unsigned*)&Ws[i];
        q[0] = pack2h(v.x, v.y);
        q[1] = pack2h(v.z, v.w);
    }
    if (tid < 128) { bsh[tid] = ba[tid]; wos[tid] = Wo[tid]; }
    __syncthreads();
    int lane = tid & 31, wave = tid >> 5;
    int n = blockIdx.x * 8 + wave;
    if (n >= n_nodes) return;
    const float* zp = Z + (size_t)n * 128;
    float a0 = 0.f, a1 = 0.f, a2 = 0.f, a3 = 0.f;
    for (int k = 0; k < 128; ++k) {
        float zv = zp[k];
        const _Float16* wr = Ws + k * 128 + lane;
        a0 += zv * (float)wr[0];
        a1 += zv * (float)wr[32];
        a2 += zv * (float)wr[64];
        a3 += zv * (float)wr[96];
    }
    float s = tanhf(a0 + bsh[lane])      * wos[lane]
            + tanhf(a1 + bsh[lane + 32]) * wos[lane + 32]
            + tanhf(a2 + bsh[lane + 64]) * wos[lane + 64]
            + tanhf(a3 + bsh[lane + 96]) * wos[lane + 96];
    #pragma unroll
    for (int off = 16; off > 0; off >>= 1) s += __shfl_xor(s, off, 32);
    if (lane == 0) atomicAdd(wout, s);
}

__global__ void han_beta(const float* __restrict__ wpart, float* __restrict__ beta, float invn) {
    if (threadIdx.x == 0 && blockIdx.x == 0) {
        float w0 = wpart[0] * invn, w1 = wpart[1] * invn;
        float m = fmaxf(w0, w1);
        float e0 = __expf(w0 - m), e1 = __expf(w1 - m);
        beta[0] = e0 / (e0 + e1); beta[1] = e1 / (e0 + e1);
        float w2 = wpart[2] * invn, w3 = wpart[3] * invn;
        float m2 = fmaxf(w2, w3);
        float e2 = __expf(w2 - m2), e3 = __expf(w3 - m2);
        beta[2] = e2 / (e2 + e3); beta[3] = e3 / (e2 + e3);
    }
}

__global__ void han_combine(const float* __restrict__ zaa, const float* __restrict__ zta,
                            const float* __restrict__ ztt, const float* __restrict__ zat,
                            const float* __restrict__ beta,
                            float* __restrict__ haout, float* __restrict__ htout, long long n) {
    long long i = (long long)blockIdx.x * blockDim.x + threadIdx.x;
    if (i < n) {
        haout[i] = beta[0] * zaa[i] + beta[1] * zta[i];
        htout[i] = beta[2] * ztt[i] + beta[3] * zat[i];
    }
}

// ---------------------------------------------------------------------------
// Fused predictor: X = [ha_out[src] | ht_out[dst]] (gathered into LDS),
// Y = relu(X @ W1 + b1) via WMMA (K=256, 128 cols), out = sigmoid(Y @ W2 + b2).
// Block tile: 64 edges. Layer-2 reduce via ds_add_f32 into LDS partials.
// ---------------------------------------------------------------------------
__global__ void han_predict(const float* __restrict__ haout, const float* __restrict__ htout,
                            const int* __restrict__ src, const int* __restrict__ dst,
                            const float* __restrict__ W1, const float* __restrict__ b1,
                            const float* __restrict__ W2, const float* __restrict__ b2,
                            float* __restrict__ out, int M) {
    __shared__ _Float16 As[64 * 34];
    __shared__ _Float16 Bs[128 * 34];
    __shared__ float part[64];
    __shared__ float w2s[128];
    __shared__ float b1s[128];

    const int tid  = threadIdx.x;
    const int lane = tid & 31;
    const int wave = tid >> 5;
    const int lo = lane >> 4;
    const int ml = lane & 15;
    const int m0 = blockIdx.x * 64;
    const int mt    = (wave >> 1) * 16;
    const int nbase = (wave & 1) * 64;

    if (tid < 128) { w2s[tid] = W2[tid]; b1s[tid] = b1[tid]; }
    if (tid < 64)  part[tid] = 0.f;

    const int ar   = tid >> 2;
    const int acg  = (tid & 3) * 8;
    const int arow = m0 + ar;
    const float amsk = (arow < M) ? 1.f : 0.f;
    const float* hap = haout;
    const float* htp = htout;
    if (arow < M) {
        hap = haout + (size_t)src[arow] * HIDN;
        htp = htout + (size_t)dst[arow] * HIDN;
    }
    const int bk  = tid >> 3;
    const int bng = (tid & 7) * 16;

    v8f acc[4] = {{}, {}, {}, {}};

    for (int k0 = 0; k0 < 256; k0 += 32) {
        {   // stage gathered A rows: 8-group never straddles the 128 split
            int kbase = k0 + acg;
            const float* base = (kbase < 128) ? (hap + kbase) : (htp + (kbase - 128));
            float4 x0 = *(const float4*)(base);
            float4 x1 = *(const float4*)(base + 4);
            unsigned* q = (unsigned*)&As[ar * 34 + acg];
            q[0] = pack2h(x0.x * amsk, x0.y * amsk);
            q[1] = pack2h(x0.z * amsk, x0.w * amsk);
            q[2] = pack2h(x1.x * amsk, x1.y * amsk);
            q[3] = pack2h(x1.z * amsk, x1.w * amsk);
        }
        {
            const float* wp = W1 + (size_t)(k0 + bk) * 128 + bng;
            float4 y0 = *(const float4*)(wp);
            float4 y1 = *(const float4*)(wp + 4);
            float4 y2 = *(const float4*)(wp + 8);
            float4 y3 = *(const float4*)(wp + 12);
            Bs[(bng +  0) * 34 + bk] = (_Float16)y0.x;
            Bs[(bng +  1) * 34 + bk] = (_Float16)y0.y;
            Bs[(bng +  2) * 34 + bk] = (_Float16)y0.z;
            Bs[(bng +  3) * 34 + bk] = (_Float16)y0.w;
            Bs[(bng +  4) * 34 + bk] = (_Float16)y1.x;
            Bs[(bng +  5) * 34 + bk] = (_Float16)y1.y;
            Bs[(bng +  6) * 34 + bk] = (_Float16)y1.z;
            Bs[(bng +  7) * 34 + bk] = (_Float16)y1.w;
            Bs[(bng +  8) * 34 + bk] = (_Float16)y2.x;
            Bs[(bng +  9) * 34 + bk] = (_Float16)y2.y;
            Bs[(bng + 10) * 34 + bk] = (_Float16)y2.z;
            Bs[(bng + 11) * 34 + bk] = (_Float16)y2.w;
            Bs[(bng + 12) * 34 + bk] = (_Float16)y3.x;
            Bs[(bng + 13) * 34 + bk] = (_Float16)y3.y;
            Bs[(bng + 14) * 34 + bk] = (_Float16)y3.z;
            Bs[(bng + 15) * 34 + bk] = (_Float16)y3.w;
        }
        __syncthreads();

        union { v16h v; _Float16 h[16]; } af;
        #pragma unroll
        for (int e = 0; e < 16; ++e) {
            int kk = ((e >> 3) << 4) + (lo << 3) + (e & 7);
            af.h[e] = As[(mt + ml) * 34 + kk];
        }
        #pragma unroll
        for (int j = 0; j < 4; ++j) {
            union { v16h v; _Float16 h[16]; } bf;
            int ncol = nbase + 16 * j + ml;
            #pragma unroll
            for (int e = 0; e < 16; ++e) {
                int kk = ((e >> 3) << 4) + (lo << 3) + (e & 7);
                bf.h[e] = Bs[ncol * 34 + kk];
            }
            acc[j] = __builtin_amdgcn_wmma_f32_16x16x32_f16(false, af.v, false, bf.v, (short)0, acc[j], false, false);
        }
        __syncthreads();
    }

    // layer 2: relu + dot with W2, accumulated per-row in LDS (ds_add_f32)
    #pragma unroll
    for (int j = 0; j < 4; ++j) {
        int col = nbase + 16 * j + ml;
        float w2v = w2s[col];
        float bv  = b1s[col];
        #pragma unroll
        for (int r = 0; r < 8; ++r) {
            float v = fmaxf(acc[j][r] + bv, 0.f);
            atomicAdd(&part[mt + r + 8 * lo], v * w2v);
        }
    }
    __syncthreads();
    if (tid < 64) {
        int row = m0 + tid;
        if (row < M) out[row] = 1.f / (1.f + __expf(-(part[tid] + b2[0])));
    }
}

// ---------------------------------------------------------------------------
// host launcher
// ---------------------------------------------------------------------------
extern "C" void kernel_launch(void* const* d_in, const int* in_sizes, int n_in,
                              void* d_out, int out_size, void* d_ws, size_t ws_size,
                              hipStream_t stream) {
    (void)in_sizes; (void)n_in; (void)out_size; (void)ws_size;

    char* ws = (char*)d_ws;
    size_t off = 0;
    auto wsalloc = [&](size_t bytes) -> char* {
        char* p = ws + off;
        off = (off + bytes + 255) & ~(size_t)255;
        return p;
    };
    float*    HA    = (float*)wsalloc((size_t)NN * HANIN * 4);
    float*    HT    = (float*)wsalloc((size_t)NN * HANIN * 4);
    float*    FS    = (float*)wsalloc((size_t)NN * HIDN * 4);
    float*    FD    = (float*)wsalloc((size_t)NN * HIDN * 4);
    float*    Zb[4];
    for (int i = 0; i < 4; ++i) Zb[i] = (float*)wsalloc((size_t)NN * HIDN * 4);
    float*    LOG   = (float*)wsalloc((size_t)EE * HH * 4);
    unsigned* MAXB  = (unsigned*)wsalloc((size_t)NN * HH * 4);
    float*    SUMB  = (float*)wsalloc((size_t)NN * HH * 4);
    float*    WPART = (float*)wsalloc(16 * 4);
    float*    BETA  = (float*)wsalloc(16 * 4);
    float*    HAOUT = FS;   // FS/FD dead after edge phase -> reuse
    float*    HTOUT = FD;

    const float* protein = (const float*)d_in[0];
    const float* genomic = (const float*)d_in[1];
    const float* W1  = (const float*)d_in[2];
    const float* b1  = (const float*)d_in[3];
    const float* W2  = (const float*)d_in[4];
    const float* b2  = (const float*)d_in[5];
    const float* gWs = (const float*)d_in[6];
    const float* gbs = (const float*)d_in[7];
    const float* gWd = (const float*)d_in[8];
    const float* gbd = (const float*)d_in[9];
    const float* gat = (const float*)d_in[10];
    const float* pW1 = (const float*)d_in[11];
    const float* pb1 = (const float*)d_in[12];
    const float* pW2 = (const float*)d_in[13];
    const float* gW1a = (const float*)d_in[14];
    const float* gb1a = (const float*)d_in[15];
    const float* gW2a = (const float*)d_in[16];
    const float* prW1 = (const float*)d_in[17];
    const float* prb1 = (const float*)d_in[18];
    const float* prW2 = (const float*)d_in[19];
    const float* prb2 = (const float*)d_in[20];
    const int* eptr[4] = {(const int*)d_in[21], (const int*)d_in[22],
                          (const int*)d_in[23], (const int*)d_in[24]};
    const int* pos_e = (const int*)d_in[25];
    const int* neg_e = (const int*)d_in[26];
    float* outp = (float*)d_out;

    auto gemm = [&](const float* A, const float* B, const float* bias, float* C,
                    int M, int K, int Nc) {
        dim3 grid((M + 63) / 64, Nc / 64);
        han_gemm_wmma<<<grid, 256, 0, stream>>>(A, B, bias, C, M, K, Nc);
    };
    auto fill = [&](void* p, unsigned v, long long n) {
        int blocks = (int)((n + 255) / 256);
        han_fill_u32<<<blocks, 256, 0, stream>>>((unsigned*)p, v, n);
    };

    // 0) zero accumulation buffers
    for (int i = 0; i < 4; ++i) fill(Zb[i], 0u, (long long)NN * HIDN);
    fill(WPART, 0u, 16);

    // 1) input projections (no relu: relu_after_w=False branch)
    gemm(protein, W2, b2, HA, NN, 1024, HANIN);
    gemm(genomic, W1, b1, HT, NN, 512,  HANIN);

    // 2) four GATv2 metapaths: (src feats, dst feats)
    const float* hs_arr[4] = {HA, HT, HT, HA};
    const float* hd_arr[4] = {HA, HA, HT, HT};
    for (int mp = 0; mp < 4; ++mp) {
        const float* Ws = gWs + (size_t)mp * HANIN * HIDN;
        const float* Wd = gWd + (size_t)mp * HANIN * HIDN;
        const float* bs = gbs + (size_t)mp * HIDN;
        const float* bd = gbd + (size_t)mp * HIDN;
        const float* at = gat + (size_t)mp * HH * DD;
        const int* src = eptr[mp];
        const int* dst = eptr[mp] + EE;

        gemm(hs_arr[mp], Ws, bs, FS, NN, HANIN, HIDN);
        gemm(hd_arr[mp], Wd, bd, FD, NN, HANIN, HIDN);

        fill(MAXB, 0x007FFFFFu, (long long)NN * HH);  // encoded -inf
        fill(SUMB, 0u,          (long long)NN * HH);

        han_edge_logits<<<EE / 8, 256, 0, stream>>>(FS, FD, at, src, dst, LOG, MAXB, EE);
        han_edge_exp<<<(EE * HH + 255) / 256, 256, 0, stream>>>(LOG, dst, MAXB, SUMB, EE);
        han_edge_scatter<<<(int)(((long long)EE * HIDN + 255) / 256), 256, 0, stream>>>(
            FS, LOG, SUMB, src, dst, Zb[mp], EE);
        han_relu_inplace<<<(int)(((long long)NN * HIDN + 255) / 256), 256, 0, stream>>>(
            Zb[mp], (long long)NN * HIDN);
    }

    // 3) semantic attention
    int sgrid = (NN + 7) / 8;
    han_sem_score<<<sgrid, 256, 0, stream>>>(Zb[0], pW1,  pb1,  pW2,  WPART + 0, NN);
    han_sem_score<<<sgrid, 256, 0, stream>>>(Zb[1], pW1,  pb1,  pW2,  WPART + 1, NN);
    han_sem_score<<<sgrid, 256, 0, stream>>>(Zb[2], gW1a, gb1a, gW2a, WPART + 2, NN);
    han_sem_score<<<sgrid, 256, 0, stream>>>(Zb[3], gW1a, gb1a, gW2a, WPART + 3, NN);
    han_beta<<<1, 64, 0, stream>>>(WPART, BETA, 1.0f / (float)NN);
    han_combine<<<(int)(((long long)NN * HIDN + 255) / 256), 256, 0, stream>>>(
        Zb[0], Zb[1], Zb[2], Zb[3], BETA, HAOUT, HTOUT, (long long)NN * HIDN);

    // 4) predictor (pos then neg, concatenated in d_out)
    han_predict<<<(EPN + 63) / 64, 256, 0, stream>>>(
        HAOUT, HTOUT, pos_e, pos_e + EPN, prW1, prb1, prW2, prb2, outp, EPN);
    han_predict<<<(EPN + 63) / 64, 256, 0, stream>>>(
        HAOUT, HTOUT, neg_e, neg_e + EPN, prW1, prb1, prW2, prb2, outp + EPN, EPN);
}
